// PatchedFalconAttention_79096117723842
// MI455X (gfx1250) — matrix operand order repaired and checked by reading, not compiled, not run on hardware
//
#include <hip/hip_runtime.h>

// ---------------------------------------------------------------------------
// Types for CDNA5 WMMA (wave32)
// ---------------------------------------------------------------------------
typedef __attribute__((ext_vector_type(16))) __bf16        v16bf;
typedef __attribute__((ext_vector_type(8)))  float         v8f;
typedef __attribute__((ext_vector_type(8)))  unsigned int  v8u;

constexpr int   NB    = 4;             // batch
constexpr int   MAXQ  = 512;
constexpr int   NH    = 71;            // heads
constexpr int   HD    = 64;            // head dim
constexpr int   HID   = 4544;          // hidden = NH*HD
constexpr int   NQKV  = (NH + 2) * HD; // 4672
constexpr int   TT    = NB * MAXQ;     // 2048 tokens
constexpr float SCALE = 0.125f;        // 1/sqrt(64)

__device__ __forceinline__ unsigned short f32_to_bf16(float f) {
  unsigned int u = __builtin_bit_cast(unsigned int, f);
  u += 0x7FFFu + ((u >> 16) & 1u);     // round-to-nearest-even
  return (unsigned short)(u >> 16);
}

__device__ __forceinline__ v8f wmma_bf16(v16bf a, v16bf b, v8f c) {
  return __builtin_amdgcn_wmma_f32_16x16x32_bf16(
      false, a, false, b, (short)0, c, false, false);
}

// ---------------------------------------------------------------------------
// CDNA5 async global -> LDS (16B per lane, tracked by ASYNCcnt)
// Generic pointer to __shared__ has the LDS byte offset in its low 32 bits.
// ---------------------------------------------------------------------------
__device__ __forceinline__ void async_load_b128(const unsigned short* lds_dst,
                                                const unsigned short* gsrc) {
  unsigned loff = (unsigned)(unsigned long long)lds_dst;
  asm volatile("global_load_async_to_lds_b128 %0, %1, off"
               :: "v"(loff), "v"(gsrc)
               : "memory");
}
__device__ __forceinline__ void wait_async0() {
  asm volatile("s_wait_asynccnt 0x0" ::: "memory");
}

// ---------------------------------------------------------------------------
// fp32 -> bf16 conversion, 4 elems/thread (n must be divisible by 4)
// ---------------------------------------------------------------------------
__global__ __launch_bounds__(256) void cvt_bf16_kernel(const float4* __restrict__ in,
                                                       unsigned long long* __restrict__ out,
                                                       int n4) {
  for (int i = blockIdx.x * 256 + threadIdx.x; i < n4; i += gridDim.x * 256) {
    float4 v = in[i];
    unsigned long long r =
        (unsigned long long)f32_to_bf16(v.x)
      | ((unsigned long long)f32_to_bf16(v.y) << 16)
      | ((unsigned long long)f32_to_bf16(v.z) << 32)
      | ((unsigned long long)f32_to_bf16(v.w) << 48);
    out[i] = r;
  }
}

// ---------------------------------------------------------------------------
// C[M][N] = A[M][K](bf16) * W[N][K](bf16)^T, fp32 accumulate via WMMA.
// Block tile 128x64, 8 waves (4 M x 2 N), wave tile 32x32 (4 WMMA / K-step).
// Tiles staged with async global->LDS, double buffered (ASYNCcnt).
// ---------------------------------------------------------------------------
constexpr int BM = 128, BN = 64, BK = 32;

__global__ __launch_bounds__(256) void gemm_bf16_wmma_kernel(
    const unsigned short* __restrict__ A,   // [M][K]
    const unsigned short* __restrict__ W,   // [N][K]
    float* __restrict__ C,                  // [M][N]
    int M, int N, int K) {
  __shared__ unsigned short lA[2][BM * BK];   // 2 x 8 KB
  __shared__ unsigned short lW[2][BN * BK];   // 2 x 4 KB

  const int tid  = threadIdx.x;
  const int lane = tid & 31;
  const int wave = tid >> 5;
  const int wm   = (wave >> 1) * 32;
  const int wn   = (wave & 1) * 32;
  const int m0   = blockIdx.y * BM;
  const int n0   = blockIdx.x * BN;

  const int a_kb = (lane >> 4) * 16;    // A-frag K base (0 / 16)
  const int b_kb = (lane >> 4) * 16;    // B-frag K base (0 / 16)
  const int l15  = lane & 15;

  v8f acc[2][2] = {};

  // issue one K-step's tiles as async b128 copies (3 per thread)
  auto issue = [&](int buf, int k0) {
#pragma unroll
    for (int t = 0; t < 2; ++t) {               // A: 128 rows x 4 chunks of 8
      int i = tid + t * 256;
      int r = i >> 2, c = (i & 3) * 8;
      async_load_b128(&lA[buf][r * BK + c],
                      A + (size_t)(m0 + r) * K + k0 + c);
    }
    {                                            // W: 64 rows x 4 chunks of 8
      int r = tid >> 2, c = (tid & 3) * 8;
      async_load_b128(&lW[buf][r * BK + c],
                      W + (size_t)(n0 + r) * K + k0 + c);
    }
  };

  const int nsteps = K / BK;
  issue(0, 0);

  for (int s = 0; s < nsteps; ++s) {
    const int cur = s & 1;
    wait_async0();          // my async copies (incl. cur buffer) landed
    __syncthreads();        // everyone's landed; prior compute fully drained
    if (s + 1 < nsteps) issue(cur ^ 1, (s + 1) * BK);

    v16bf af[2];
#pragma unroll
    for (int x = 0; x < 2; ++x) {
      v8u u;
#pragma unroll
      for (int p = 0; p < 8; ++p) {
        int kk = a_kb + 2 * p;
        u[p] = *(const unsigned int*)(&lA[cur][(wm + x * 16 + l15) * BK + kk]);
      }
      af[x] = __builtin_bit_cast(v16bf, u);
    }
    v16bf bfr[2];
#pragma unroll
    for (int y = 0; y < 2; ++y) {
      v8u u;
#pragma unroll
      for (int p = 0; p < 8; ++p)
        u[p] = *(const unsigned int*)(&lW[cur][(wn + y * 16 + l15) * BK + b_kb + 2 * p]);
      bfr[y] = __builtin_bit_cast(v16bf, u);
    }
#pragma unroll
    for (int x = 0; x < 2; ++x)
#pragma unroll
      for (int y = 0; y < 2; ++y)
        acc[x][y] = wmma_bf16(af[x], bfr[y], acc[x][y]);
  }

  const int mhi = (lane >> 4) * 8;   // C/D layout: VGPR r -> M = r + mhi
#pragma unroll
  for (int x = 0; x < 2; ++x)
#pragma unroll
    for (int y = 0; y < 2; ++y) {
      int col = n0 + wn + y * 16 + l15;
#pragma unroll
      for (int r = 0; r < 8; ++r) {
        int row = m0 + wm + x * 16 + r + mhi;
        C[(size_t)row * N + col] = acc[x][y][r];
      }
    }
}

// ---------------------------------------------------------------------------
// RoPE + QKV split. One block per token.
//   qbuf  [B][H][512][64] bf16, kbuf [B][512][64] bf16, vtbuf [B][64][512] bf16
// ---------------------------------------------------------------------------
__global__ __launch_bounds__(256) void rope_split_kernel(
    const float* __restrict__ fused,     // [T][4672]
    const float* __restrict__ inv_freq,  // [32]
    const int* __restrict__ pos_ids,     // [T]
    unsigned short* __restrict__ qbuf,
    unsigned short* __restrict__ kbuf,
    unsigned short* __restrict__ vtbuf) {
  const int t = blockIdx.x;
  const int b = t >> 9;
  const int i = t & (MAXQ - 1);
  const float pos = (float)pos_ids[t];
  const float* frow = fused + (size_t)t * NQKV;

  for (int idx = threadIdx.x; idx < NQKV; idx += 256) {
    const int hh = idx >> 6;
    const int d  = idx & 63;
    float x = frow[idx];
    float outv;
    if (hh <= NH) {  // rotary on q heads and the single k head
      float fr = pos * inv_freq[d & 31];
      float c, s;
      __sincosf(fr, &s, &c);
      float other = (d < 32) ? -frow[idx + 32] : frow[idx - 32];
      outv = x * c + other * s;
    } else {
      outv = x;      // v passthrough
    }
    unsigned short bf = f32_to_bf16(outv);
    if (hh < NH)
      qbuf[(((size_t)(b * NH + hh)) * MAXQ + i) * HD + d] = bf;
    else if (hh == NH)
      kbuf[((size_t)b * MAXQ + i) * HD + d] = bf;
    else
      vtbuf[((size_t)b * HD + d) * MAXQ + i] = bf;
  }
}

// ---------------------------------------------------------------------------
// Attention: one wave per (b, h, 16-row q tile).
// ---------------------------------------------------------------------------
__global__ __launch_bounds__(32) void attn_kernel(
    const unsigned short* __restrict__ qbuf,   // [B][H][512][64]
    const unsigned short* __restrict__ kbuf,   // [B][512][64]
    const unsigned short* __restrict__ vtbuf,  // [B][64][512]
    const int* __restrict__ kv_len_arr,        // [B]
    unsigned short* __restrict__ obuf) {       // [T][H*64] bf16
  __shared__ float          sc[16 * 512];
  __shared__ unsigned short pb[16 * 512];

  const int bid  = blockIdx.x;
  const int qt   = bid & 31;
  const int h    = (bid >> 5) % NH;
  const int b    = bid / (32 * NH);
  const int lane = threadIdx.x;
  const int q0   = qt * 16;
  const int kv_len = kv_len_arr[b];

  const int l15  = lane & 15;
  const int a_kb = (lane >> 4) * 16;
  const int b_kb = (lane >> 4) * 16;
  const int mhi  = (lane >> 4) * 8;

  // Q fragments (K = d: 0..31, 32..63)
  const unsigned short* qrow =
      qbuf + (((size_t)(b * NH + h)) * MAXQ + q0 + l15) * HD;
  v8u qa0u, qa1u;
#pragma unroll
  for (int p = 0; p < 8; ++p) {
    int kk = a_kb + 2 * p;
    qa0u[p] = *(const unsigned int*)(qrow + kk);
    qa1u[p] = *(const unsigned int*)(qrow + 32 + kk);
  }
  v16bf qa0 = __builtin_bit_cast(v16bf, qa0u);
  v16bf qa1 = __builtin_bit_cast(v16bf, qa1u);

  // scores: 32 key tiles of 16
  for (int n0 = 0; n0 < MAXQ; n0 += 16) {
    const int j = n0 + l15;
    const unsigned short* krow = kbuf + ((size_t)b * MAXQ + j) * HD;
    v8u kb0u, kb1u;
#pragma unroll
    for (int p = 0; p < 8; ++p) {
      kb0u[p] = *(const unsigned int*)(krow + b_kb + 2 * p);
      kb1u[p] = *(const unsigned int*)(krow + 32 + b_kb + 2 * p);
    }
    v8f acc = {};
    acc = wmma_bf16(qa0, __builtin_bit_cast(v16bf, kb0u), acc);
    acc = wmma_bf16(qa1, __builtin_bit_cast(v16bf, kb1u), acc);

#pragma unroll
    for (int r = 0; r < 8; ++r) {
      int row = r + mhi;
      int qi  = q0 + row;
      bool ok = (j <= qi) && (j < kv_len);
      sc[row * 512 + j] = ok ? acc[r] * SCALE : -1.0e30f;
    }
  }
  __syncthreads();

  // wave-wide softmax per row
  for (int row = 0; row < 16; ++row) {
    float mx = -3.0e38f;
    for (int c = lane; c < 512; c += 32) mx = fmaxf(mx, sc[row * 512 + c]);
#pragma unroll
    for (int off = 16; off; off >>= 1) mx = fmaxf(mx, __shfl_xor(mx, off, 32));
    float sum = 0.0f;
    for (int c = lane; c < 512; c += 32) {
      float e = __expf(sc[row * 512 + c] - mx);
      sc[row * 512 + c] = e;
      sum += e;
    }
#pragma unroll
    for (int off = 16; off; off >>= 1) sum += __shfl_xor(sum, off, 32);
    float inv = 1.0f / sum;
    for (int c = lane; c < 512; c += 32)
      pb[row * 512 + c] = f32_to_bf16(sc[row * 512 + c] * inv);
  }
  __syncthreads();

  // out = P(16x512) * V(512x64): load each P frag once, 4 V-tile WMMAs
  v8f oacc[4] = {};
  for (int k0 = 0; k0 < MAXQ; k0 += 32) {
    v8u pau;
#pragma unroll
    for (int p = 0; p < 8; ++p) {
      int kk = a_kb + 2 * p;
      pau[p] = *(const unsigned int*)(pb + l15 * 512 + k0 + kk);
    }
    v16bf pf = __builtin_bit_cast(v16bf, pau);
#pragma unroll
    for (int t = 0; t < 4; ++t) {
      const unsigned short* vrow = vtbuf + ((size_t)b * HD + t * 16 + l15) * MAXQ;
      v8u vbu;
#pragma unroll
      for (int p = 0; p < 8; ++p)
        vbu[p] = *(const unsigned int*)(vrow + k0 + b_kb + 2 * p);
      oacc[t] = wmma_bf16(pf, __builtin_bit_cast(v16bf, vbu), oacc[t]);
    }
  }
#pragma unroll
  for (int t = 0; t < 4; ++t)
#pragma unroll
    for (int r = 0; r < 8; ++r) {
      int row = r + mhi;
      obuf[((size_t)(b * MAXQ + q0 + row)) * HID + h * HD + t * 16 + l15] =
          f32_to_bf16(oacc[t][r]);
    }
}

// ---------------------------------------------------------------------------
// Launcher
// ---------------------------------------------------------------------------
extern "C" void kernel_launch(void* const* d_in, const int* in_sizes, int n_in,
                              void* d_out, int out_size, void* d_ws, size_t ws_size,
                              hipStream_t stream) {
  (void)in_sizes; (void)n_in; (void)out_size; (void)ws_size;
  const float* hidden   = (const float*)d_in[0];   // [T][HID]
  const float* w_qkv    = (const float*)d_in[1];   // [NQKV][HID]
  const float* w_dense  = (const float*)d_in[2];   // [HID][HID]
  const float* inv_freq = (const float*)d_in[3];   // [32]
  const int*   pos_ids  = (const int*)d_in[6];     // [T]
  const int*   kv_len   = (const int*)d_in[9];     // [B]
  float*       out      = (float*)d_out;           // [T][HID]

  constexpr size_t SZ_HID_BF  = (size_t)TT * HID * 2;    // hidden bf16 / qbuf alias
  constexpr size_t SZ_WQKV_BF = (size_t)NQKV * HID * 2;
  constexpr size_t SZ_WD_BF   = (size_t)HID * HID * 2;
  constexpr size_t SZ_FUSED   = (size_t)TT * NQKV * 4;   // fused fp32 / attn bf16 alias
  constexpr size_t SZ_KV      = (size_t)NB * MAXQ * HD * 2;

  char* base = (char*)d_ws;
  unsigned short* hid_bf  = (unsigned short*)(base);
  unsigned short* wqkv_bf = (unsigned short*)(base + SZ_HID_BF);
  unsigned short* wd_bf   = (unsigned short*)(base + SZ_HID_BF + SZ_WQKV_BF);
  float*          fused   = (float*)(base + SZ_HID_BF + SZ_WQKV_BF + SZ_WD_BF);
  unsigned short* kbuf    = (unsigned short*)(base + SZ_HID_BF + SZ_WQKV_BF + SZ_WD_BF + SZ_FUSED);
  unsigned short* vtbuf   = (unsigned short*)((char*)kbuf + SZ_KV);
  unsigned short* qbuf    = hid_bf;                 // alias: dead after gemm1
  unsigned short* attn_bf = (unsigned short*)fused; // alias: dead after rope

  cvt_bf16_kernel<<<2048, 256, 0, stream>>>((const float4*)hidden,
                                            (unsigned long long*)hid_bf, TT * HID / 4);
  cvt_bf16_kernel<<<4096, 256, 0, stream>>>((const float4*)w_qkv,
                                            (unsigned long long*)wqkv_bf, NQKV * HID / 4);
  cvt_bf16_kernel<<<4096, 256, 0, stream>>>((const float4*)w_dense,
                                            (unsigned long long*)wd_bf, HID * HID / 4);

  gemm_bf16_wmma_kernel<<<dim3(NQKV / BN, TT / BM), 256, 0, stream>>>(
      hid_bf, wqkv_bf, fused, TT, NQKV, HID);

  rope_split_kernel<<<TT, 256, 0, stream>>>(fused, inv_freq, pos_ids,
                                            qbuf, kbuf, vtbuf);

  attn_kernel<<<NB * NH * (MAXQ / 16), 32, 0, stream>>>(qbuf, kbuf, vtbuf,
                                                        kv_len, attn_bf);

  gemm_bf16_wmma_kernel<<<dim3(HID / BN, TT / BM), 256, 0, stream>>>(
      attn_bf, wd_bf, out, TT, HID, HID);
}
